// Attention_56014963474997
// MI455X (gfx1250) — matrix operand order repaired and checked by reading, not compiled, not run on hardware
//
#include <hip/hip_runtime.h>
#include <hip/hip_bf16.h>
#include <math.h>

// Problem constants (from reference)
#define B_ 32
#define S_ 4096
#define H_ 512

// Kernel-B tiling
constexpr int SROWS = 128;          // s-rows per block (8 waves x 16)
constexpr int PITCH = 520;          // bf16 elems per LDS row (512 + 8 pad => 4-bank skew)
constexpr int KT    = 32;           // K per wmma_f32_16x16x32_bf16
constexpr int CROWS = 16;           // s-rows per block in context kernel

typedef __attribute__((ext_vector_type(16))) __bf16 v16bf;
typedef __attribute__((ext_vector_type(8)))  __bf16 v8bf;
typedef __attribute__((ext_vector_type(8)))  float  v8f;

union Frag16 { v16bf v; v8bf h[2]; };

#if __has_builtin(__builtin_amdgcn_tanhf)
__device__ __forceinline__ float fast_tanh(float x) { return __builtin_amdgcn_tanhf(x); }
#else
__device__ __forceinline__ float fast_tanh(float x) { return tanhf(x); }
#endif

// ---------------------------------------------------------------------------
// Kernel A: qb[b,o] = sum_h query[b,h]*Wa_w[o,h] + Wa_b[o] + Ua_b[o]
// ---------------------------------------------------------------------------
__global__ __launch_bounds__(256) void qproj_kernel(
    const float* __restrict__ query,   // [B,1,H]
    const float* __restrict__ Wa_w,    // [H,H] row-major [o][h]
    const float* __restrict__ Wa_b,    // [H]
    const float* __restrict__ Ua_b,    // [H]
    float* __restrict__ qb)            // [B,H]
{
    __shared__ float qLds[H_];
    const int b = blockIdx.x, tid = threadIdx.x;
    for (int i = tid; i < H_; i += 256) qLds[i] = query[b * H_ + i];
    __syncthreads();
    for (int o = tid; o < H_; o += 256) {
        const float4* w4 = (const float4*)(Wa_w + (size_t)o * H_);
        const float4* q4 = (const float4*)qLds;
        float acc = 0.f;
#pragma unroll 8
        for (int c = 0; c < H_ / 4; ++c) {
            float4 w = w4[c];
            float4 q = q4[c];
            acc += w.x * q.x + w.y * q.y + w.z * q.z + w.w * q.w;
        }
        qb[b * H_ + o] = acc + Wa_b[o] + Ua_b[o];
    }
}

// ---------------------------------------------------------------------------
// Kernel B: scores[b,s] = Va_w . tanh(qb[b,:] + keys[b,s,:] @ Ua_w^T) + Va_b
// One block = one batch x 128 s-rows.  Keys tile staged once to LDS as bf16,
// Ua_w staged per 16-col o-tile (global loads pipelined across the barrier).
// Each wave owns a 16-row stripe; fragment LDS loads are double-buffered so
// each v_wmma_f32_16x16x32_bf16 depends only on loads issued one step ahead.
// ---------------------------------------------------------------------------
__global__ __launch_bounds__(256) void bahdanau_scores_wmma(
    const float* __restrict__ keys,    // [B,S,H]
    const float* __restrict__ Ua_w,    // [H,H] row-major [o][h]
    const float* __restrict__ qb,      // [B,H]  (q_proj + Wa_b + Ua_b)
    const float* __restrict__ Va_w,    // [H]
    const float* __restrict__ Va_b,    // [1]
    float* __restrict__ scores)        // [B,S]
{
    __shared__ __bf16 keysLds[SROWS * PITCH];   // 133,120 B
    __shared__ __bf16 uaLds[16 * PITCH];        //  16,640 B

    const int b    = blockIdx.y;
    const int s0   = blockIdx.x * SROWS;
    const int tid  = threadIdx.x;
    const int lane = tid & 31;
    const int wave = tid >> 5;
    const int n    = lane & 15;   // column within tile (N index)
    const int hi   = lane >> 4;   // half-wave selector

    // --- stage keys tile (128 x 512) f32 -> bf16 into LDS, read HBM once ---
    const float* kbase = keys + ((size_t)b * S_ + s0) * H_;
    for (int idx = tid; idx < SROWS * (H_ / 4); idx += 256) {
        const int row = idx >> 7;        // 128 float4 per row
        const int c4  = idx & 127;
        float4 f = ((const float4*)(kbase + (size_t)row * H_))[c4];
        __bf16* dst = &keysLds[row * PITCH + c4 * 4];
        dst[0] = (__bf16)f.x; dst[1] = (__bf16)f.y;
        dst[2] = (__bf16)f.z; dst[3] = (__bf16)f.w;
    }

    float scoreAcc[8];
#pragma unroll
    for (int r = 0; r < 8; ++r) scoreAcc[r] = 0.f;

    const float vab = Va_b[0];
    const __bf16* arow = &keysLds[(wave * 16 + n) * PITCH];
    const __bf16* brow = &uaLds[n * PITCH];

    // Pipelined Ua_w staging registers: 8 float4 per thread per o-tile.
    // idx = j*256 + tid ; row = idx>>7 ; c4 = idx&127
    float4 stage[8];
#pragma unroll
    for (int j = 0; j < 8; ++j) {
        const int idx = j * 256 + tid;
        stage[j] = ((const float4*)(Ua_w + (size_t)(idx >> 7) * H_))[idx & 127];
    }

    for (int ot = 0; ot < H_ / 16; ++ot) {
        const int o0 = ot * 16;
        __syncthreads();   // uaLds WAR + (first iter) keys staging done
        // commit staged Ua_w rows [o0, o0+16) as bf16
#pragma unroll
        for (int j = 0; j < 8; ++j) {
            const int idx = j * 256 + tid;
            const float4 f = stage[j];
            __bf16* dst = &uaLds[(idx >> 7) * PITCH + (idx & 127) * 4];
            dst[0] = (__bf16)f.x; dst[1] = (__bf16)f.y;
            dst[2] = (__bf16)f.z; dst[3] = (__bf16)f.w;
        }
        __syncthreads();

        // issue next o-tile's global loads now; latency hides behind WMMAs
        if (ot + 1 < H_ / 16) {
#pragma unroll
            for (int j = 0; j < 8; ++j) {
                const int idx = j * 256 + tid;
                stage[j] = ((const float4*)(Ua_w +
                    (size_t)(o0 + 16 + (idx >> 7)) * H_))[idx & 127];
            }
        }

        v8f acc = {};
        // distance-1 double-buffered fragment pipeline
        Frag16 a0, b0, a1, b1;
        a0.h[0] = *(const v8bf*)(arow + hi * 8);
        a0.h[1] = *(const v8bf*)(arow + 16 + hi * 8);
        b0.h[0] = *(const v8bf*)(brow + hi * 8);
        b0.h[1] = *(const v8bf*)(brow + 16 + hi * 8);
#pragma unroll
        for (int kt = 0; kt < H_ / KT; ++kt) {
            if (kt + 1 < H_ / KT) {
                const int h1 = (kt + 1) * KT;
                // A (16x32 bf16): lanes 0-15 hold K{0-7,16-23}, lanes 16-31 K{8-15,24-31}
                a1.h[0] = *(const v8bf*)(arow + h1 + hi * 8);
                a1.h[1] = *(const v8bf*)(arow + h1 + 16 + hi * 8);
                // B (32x16 bf16): lane = N, lanes 0-15 hold K{0-7,16-23}, lanes 16-31 K{8-15,24-31}
                b1.h[0] = *(const v8bf*)(brow + h1 + hi * 8);
                b1.h[1] = *(const v8bf*)(brow + h1 + 16 + hi * 8);
            }
            acc = __builtin_amdgcn_wmma_f32_16x16x32_bf16(
                false, a0.v, false, b0.v, (short)0, acc, false, false);
            a0 = a1; b0 = b1;
        }

        // Fused epilogue: + bias, tanh (hw V_TANH_F32), * Va_w.
        // The 16-column reduction is linear -> defer butterfly to the end.
        const float qbias = qb[b * H_ + o0 + n];
        const float vaw   = Va_w[o0 + n];
#pragma unroll
        for (int r = 0; r < 8; ++r)
            scoreAcc[r] += fast_tanh(acc[r] + qbias) * vaw;
    }

    // Single butterfly reduction across the 16-lane column group.
    // D layout: VGPR r -> row (r + 8*hi).
#pragma unroll
    for (int r = 0; r < 8; ++r) {
        float p = scoreAcc[r];
        p += __shfl_xor(p, 1);
        p += __shfl_xor(p, 2);
        p += __shfl_xor(p, 4);
        p += __shfl_xor(p, 8);
        if (n == 0) {
            const int s = s0 + wave * 16 + hi * 8 + r;
            scores[b * S_ + s] = p + vab;
        }
    }
}

// ---------------------------------------------------------------------------
// Kernel C1: per-batch softmax statistics (max, 1/sum(exp))
// ---------------------------------------------------------------------------
__global__ __launch_bounds__(256) void softmax_stats(
    const float* __restrict__ scores,  // [B,S]
    float* __restrict__ stats)         // [B,2] = {max, inv_sum}
{
    __shared__ float red[256];
    const int b = blockIdx.x, tid = threadIdx.x;
    const float* sc = scores + b * S_;

    float m = -INFINITY;
    for (int i = tid; i < S_; i += 256) m = fmaxf(m, sc[i]);
    red[tid] = m; __syncthreads();
    for (int off = 128; off > 0; off >>= 1) {
        if (tid < off) red[tid] = fmaxf(red[tid], red[tid + off]);
        __syncthreads();
    }
    m = red[0]; __syncthreads();

    float sum = 0.f;
    for (int i = tid; i < S_; i += 256) sum += __expf(sc[i] - m);
    red[tid] = sum; __syncthreads();
    for (int off = 128; off > 0; off >>= 1) {
        if (tid < off) red[tid] += red[tid + off];
        __syncthreads();
    }
    if (tid == 0) { stats[b * 2] = m; stats[b * 2 + 1] = 1.f / red[0]; }
}

// ---------------------------------------------------------------------------
// Kernel C2: context[b,s,:] = softmax_weight(b,s) * values[b,:]  (256 MB stream)
// ---------------------------------------------------------------------------
__global__ __launch_bounds__(256) void context_kernel(
    const float* __restrict__ scores,  // [B,S]
    const float* __restrict__ stats,   // [B,2]
    const float* __restrict__ values,  // [B,1,H]
    float* __restrict__ out)           // [B,S,H]
{
    const int b   = blockIdx.y;
    const int s0  = blockIdx.x * CROWS;
    const int tid = threadIdx.x;
    const int c4  = tid & 127;   // float4 column (128 per row)
    const int r0  = tid >> 7;    // 0 or 1

    const float m   = stats[b * 2];
    const float inv = stats[b * 2 + 1];
    const float4 v  = ((const float4*)(values + b * H_))[c4];
    float4* out4 = (float4*)out;

#pragma unroll
    for (int rr = 0; rr < CROWS; rr += 2) {
        const int s = s0 + rr + r0;
        const float w = __expf(scores[b * S_ + s] - m) * inv;
        float4 o = make_float4(w * v.x, w * v.y, w * v.z, w * v.w);
        out4[(size_t)(b * S_ + s) * (H_ / 4) + c4] = o;
    }
}

// ---------------------------------------------------------------------------
extern "C" void kernel_launch(void* const* d_in, const int* in_sizes, int n_in,
                              void* d_out, int out_size, void* d_ws, size_t ws_size,
                              hipStream_t stream) {
    const float* query  = (const float*)d_in[0];
    const float* keys   = (const float*)d_in[1];
    const float* values = (const float*)d_in[2];
    const float* Wa_w   = (const float*)d_in[3];
    const float* Wa_b   = (const float*)d_in[4];
    const float* Ua_w   = (const float*)d_in[5];
    const float* Ua_b   = (const float*)d_in[6];
    const float* Va_w   = (const float*)d_in[7];
    const float* Va_b   = (const float*)d_in[8];
    float* out = (float*)d_out;

    // Workspace layout (f32): qb[B*H] | scores[B*S] | stats[B*2]  (~590 KB)
    float* qb     = (float*)d_ws;
    float* scores = qb + B_ * H_;
    float* stats  = scores + B_ * S_;

    qproj_kernel<<<B_, 256, 0, stream>>>(query, Wa_w, Wa_b, Ua_b, qb);
    bahdanau_scores_wmma<<<dim3(S_ / SROWS, B_), 256, 0, stream>>>(
        keys, Ua_w, qb, Va_w, Va_b, scores);
    softmax_stats<<<B_, 256, 0, stream>>>(scores, stats);
    context_kernel<<<dim3(S_ / CROWS, B_), 256, 0, stream>>>(
        scores, stats, values, out);
}
